// GQA_15719580303902
// MI455X (gfx1250) — compile-verified, matrix operand order checked by reading, not run on hardware
//
#include <hip/hip_runtime.h>
#include <hip/hip_bf16.h>

typedef unsigned short u16;
typedef unsigned int u32;
typedef __attribute__((ext_vector_type(16))) __bf16 v16bf;
typedef __attribute__((ext_vector_type(8)))  float  v8f;

union FragU { uint4 q[2]; v16bf v; };

__device__ __forceinline__ u16 f2bf(float f) {
    union { float f; unsigned u; } c; c.f = f;
    unsigned r = c.u + 0x7FFFu + ((c.u >> 16) & 1u);   // RNE
    return (u16)(r >> 16);
}
__device__ __forceinline__ u32 pack2bf(float lo, float hi) {
    return (u32)f2bf(lo) | ((u32)f2bf(hi) << 16);
}

// ---------------------------------------------------------------- convert
__global__ void cvt_f32_bf16(const float* __restrict__ in, u16* __restrict__ out, int n) {
    int i = blockIdx.x * blockDim.x + threadIdx.x;
    int s = gridDim.x * blockDim.x;
    for (; i < n; i += s) out[i] = f2bf(in[i]);
}

// ---------------------------------------------------------------- GEMM: C[M,N] = A[M,K] @ W[N,K]^T
// 128x128 tile, BK=32, 256 threads = 8 waves (4x2), wave patch 32x64.
// Register-prefetch pipeline: tile k+1 global loads issue before tile-k WMMAs.
template <int OUTF32>
__global__ __launch_bounds__(256) void gemm_xwt(const u16* __restrict__ A,
                                                const u16* __restrict__ W,
                                                float* __restrict__ Cf,
                                                u16* __restrict__ Cb,
                                                int M, int N, int K) {
    __shared__ u16 As[128 * 40];
    __shared__ u16 Bs[128 * 40];

    const int tid  = threadIdx.x;
    const int lane = tid & 31, wave = tid >> 5;
    const int wm = wave >> 1, wn = wave & 1;
    const int half = lane >> 4, ln = lane & 15;

    const long rowA0 = (long)blockIdx.y * 128;
    const long rowB0 = (long)blockIdx.x * 128;

    v8f acc[2][4];
#pragma unroll
    for (int i = 0; i < 2; ++i)
#pragma unroll
        for (int j = 0; j < 4; ++j) acc[i][j] = (v8f){};

    const int sr = tid >> 1, sc = (tid & 1) * 16;
    const u16* ag = A + (rowA0 + sr) * (long)K + sc;
    const u16* bg = W + (rowB0 + sr) * (long)K + sc;
    u16* asd = &As[sr * 40 + sc];
    u16* bsd = &Bs[sr * 40 + sc];

    uint4 pa0 = *(const uint4*)(ag);
    uint4 pa1 = *(const uint4*)(ag + 8);
    uint4 pb0 = *(const uint4*)(bg);
    uint4 pb1 = *(const uint4*)(bg + 8);

    for (int k0 = 0; k0 < K; k0 += 32) {
        __syncthreads();
        *(uint4*)(asd)     = pa0;
        *(uint4*)(asd + 8) = pa1;
        *(uint4*)(bsd)     = pb0;
        *(uint4*)(bsd + 8) = pb1;
        __syncthreads();

        if (k0 + 32 < K) {             // prefetch next tile; latency hides under WMMAs
            ag += 32; bg += 32;
            pa0 = *(const uint4*)(ag);
            pa1 = *(const uint4*)(ag + 8);
            pb0 = *(const uint4*)(bg);
            pb1 = *(const uint4*)(bg + 8);
        }

        FragU a[2], b[4];
#pragma unroll
        for (int mt = 0; mt < 2; ++mt) {
            int base = (wm * 32 + mt * 16 + ln) * 40 + half * 8;
            a[mt].q[0] = *(const uint4*)&As[base];
            a[mt].q[1] = *(const uint4*)&As[base + 16];
        }
#pragma unroll
        for (int nt = 0; nt < 4; ++nt) {
            int base = (wn * 64 + nt * 16 + ln) * 40 + half * 8;
            b[nt].q[0] = *(const uint4*)&Bs[base];
            b[nt].q[1] = *(const uint4*)&Bs[base + 16];
        }
#pragma unroll
        for (int mt = 0; mt < 2; ++mt)
#pragma unroll
            for (int nt = 0; nt < 4; ++nt)
                acc[mt][nt] = __builtin_amdgcn_wmma_f32_16x16x32_bf16(
                    false, a[mt].v, false, b[nt].v, (short)0, acc[mt][nt], false, false);
    }

#pragma unroll
    for (int mt = 0; mt < 2; ++mt)
#pragma unroll
        for (int nt = 0; nt < 4; ++nt)
#pragma unroll
            for (int v = 0; v < 8; ++v) {
                long r = rowA0 + wm * 32 + mt * 16 + v + 8 * half;
                long c = rowB0 + wn * 64 + nt * 16 + ln;
                float val = acc[mt][nt][v];
                if (OUTF32) Cf[r * (long)N + c] = val;
                else        Cb[r * (long)N + c] = f2bf(val);
            }
}

// ---------------------------------------------------------------- flash attention (sliding window + sink)
// Transposed-score formulation: S^T = K @ Q^T, O^T = V^T @ P^T.
// Lane dim = query -> per-lane m/l scalars, 1 shfl per reduction, vectorized P/Y writes.
// QKV packed: row stride 3072, Q at col 0, K at 2048, V at 2560.
// grid: B * NH * (T/64); block: 128 threads = 4 waves, 16 query rows each; 64-key chunks.
__global__ __launch_bounds__(128) void attn_fwd(const u16* __restrict__ QKV,
                                                u16* __restrict__ Y) {
    constexpr int T = 2048, CC = 2048, QKVC = 3072, HS = 128;
    constexpr int KOFF = 2048, VOFF = 2560;
    __shared__ u16 Ks[64 * 136];     // K chunk row-major (key, hs)        17408 B
    __shared__ u16 Vt[128 * 72];     // V chunk transposed (hs, key)       18432 B
    __shared__ u16 Ps[4 * 16 * 72];  // per-wave P^T patch (query, key)     9216 B

    const int qb = blockIdx.x & 31;
    const int h  = (blockIdx.x >> 5) & 15;
    const int b  = blockIdx.x >> 9;
    const int kv = h >> 2;

    const int tid  = threadIdx.x;
    const int lane = tid & 31, wave = tid >> 5;
    const int half = lane >> 4, ln = lane & 15;
    const int qlo  = qb * 64;
    const int iq   = qlo + wave * 16 + ln;   // this lane's query index

    // Q fragments (B operand of S^T): lane = query, contiguous hs runs
    FragU qf[4];
    {
        const u16* qp = QKV + ((long)(b * T + iq)) * QKVC + h * HS + half * 8;
#pragma unroll
        for (int kc = 0; kc < 4; ++kc) {
            qf[kc].q[0] = *(const uint4*)(qp + kc * 32);
            qf[kc].q[1] = *(const uint4*)(qp + kc * 32 + 16);
        }
    }

    v8f o[8];                                 // O^T: 8 hs tiles x 16 queries
#pragma unroll
    for (int i = 0; i < 8; ++i) o[i] = (v8f){};
    float m = -1e30f, l = 0.0f;               // per-lane (per-query) stats

    u16* Pw = &Ps[wave * 16 * 72];
    const int skr = tid >> 1, skc = tid & 1;               // K staging: row 0..63, col half
    const int svr = (tid >> 3) * 4, svc = (tid & 7) * 16;  // V staging: 4 rows x 16 cols
    const float scale = 0.08838834764831845f;  // 1/sqrt(128)

    for (int c = 0; c < qlo + 64; c += 64) {
        if (c >= 64 && (c + 63) < (qlo - 1023)) continue;  // outside window & sink (uniform)

        __syncthreads();
        {   // stage K chunk: 64 x 128 row-major
            const u16* kp = QKV + ((long)(b * T + c + skr)) * QKVC + KOFF + kv * HS + skc * 64;
            u16* kd = &Ks[skr * 136 + skc * 64];
#pragma unroll
            for (int i = 0; i < 8; ++i) *(uint4*)(kd + i * 8) = *(const uint4*)(kp + i * 8);
            // stage V transposed: Vt[hs][key], 4 keys packed per b64 store
            union { uint4 q[2]; u16 u[16]; } row[4];
#pragma unroll
            for (int r = 0; r < 4; ++r) {
                const u16* vp = QKV + ((long)(b * T + c + svr + r)) * QKVC + VOFF + kv * HS + svc;
                row[r].q[0] = *(const uint4*)(vp);
                row[r].q[1] = *(const uint4*)(vp + 8);
            }
#pragma unroll
            for (int cc = 0; cc < 16; ++cc) {
                uint2 pk;
                pk.x = (u32)row[0].u[cc] | ((u32)row[1].u[cc] << 16);
                pk.y = (u32)row[2].u[cc] | ((u32)row[3].u[cc] << 16);
                *(uint2*)&Vt[(svc + cc) * 72 + svr] = pk;
            }
        }
        __syncthreads();

        // S^T = K @ Q^T : 64 keys (4 M-tiles) x 16 queries
        v8f s[4];
#pragma unroll
        for (int t = 0; t < 4; ++t) s[t] = (v8f){};
#pragma unroll
        for (int kc = 0; kc < 4; ++kc) {
#pragma unroll
            for (int t = 0; t < 4; ++t) {
                FragU kf;
                int base = (t * 16 + ln) * 136 + kc * 32 + half * 8;
                kf.q[0] = *(const uint4*)&Ks[base];
                kf.q[1] = *(const uint4*)&Ks[base + 16];
                s[t] = __builtin_amdgcn_wmma_f32_16x16x32_bf16(
                    false, kf.v, false, qf[kc].v, (short)0, s[t], false, false);
            }
        }

        // masked scores; key j = c + t*16 + 8*half + v lies along registers
        float a[4][8];
#pragma unroll
        for (int t = 0; t < 4; ++t)
#pragma unroll
            for (int v = 0; v < 8; ++v) {
                int j = c + t * 16 + 8 * half + v;
                float av = s[t][v] * scale;
                bool ok = (j <= iq) && (j >= iq - 1023 || j < 4);
                a[t][v] = ok ? av : -1e30f;
            }

        // online softmax: pairwise-tree reductions + one cross-half shuffle
        float tm[4];
#pragma unroll
        for (int t = 0; t < 4; ++t)
            tm[t] = fmaxf(fmaxf(fmaxf(a[t][0], a[t][1]), fmaxf(a[t][2], a[t][3])),
                          fmaxf(fmaxf(a[t][4], a[t][5]), fmaxf(a[t][6], a[t][7])));
        float rm = fmaxf(fmaxf(tm[0], tm[1]), fmaxf(tm[2], tm[3]));
        rm = fmaxf(rm, __shfl_xor(rm, 16));
        float mn   = fmaxf(m, rm);
        float corr = __expf(m - mn);
        float p[4][8];
#pragma unroll
        for (int t = 0; t < 4; ++t)
#pragma unroll
            for (int v = 0; v < 8; ++v) p[t][v] = __expf(a[t][v] - mn);
        float ts[4];
#pragma unroll
        for (int t = 0; t < 4; ++t)
            ts[t] = ((p[t][0] + p[t][1]) + (p[t][2] + p[t][3])) +
                    ((p[t][4] + p[t][5]) + (p[t][6] + p[t][7]));
        float rs = (ts[0] + ts[1]) + (ts[2] + ts[3]);
        rs += __shfl_xor(rs, 16);
        l = l * corr + rs;
        m = mn;
#pragma unroll
        for (int nt = 0; nt < 8; ++nt)
#pragma unroll
            for (int v = 0; v < 8; ++v) o[nt][v] = o[nt][v] * corr;

        // P^T -> LDS: 8 consecutive keys per lane per tile -> packed b128 stores
#pragma unroll
        for (int t = 0; t < 4; ++t) {
            uint4 w;
            w.x = pack2bf(p[t][0], p[t][1]);
            w.y = pack2bf(p[t][2], p[t][3]);
            w.z = pack2bf(p[t][4], p[t][5]);
            w.w = pack2bf(p[t][6], p[t][7]);
            *(uint4*)&Pw[ln * 72 + t * 16 + 8 * half] = w;
        }

        // P^T B-fragments (lane = query, contiguous key runs)
        FragU pf[2];
#pragma unroll
        for (int kc2 = 0; kc2 < 2; ++kc2) {
            int pb = ln * 72 + kc2 * 32 + half * 8;
            pf[kc2].q[0] = *(const uint4*)&Pw[pb];
            pf[kc2].q[1] = *(const uint4*)&Pw[pb + 16];
        }

        // O^T += V^T @ P^T : 8 hs tiles x 2 key-subchunks
#pragma unroll
        for (int nt = 0; nt < 8; ++nt) {
#pragma unroll
            for (int kc2 = 0; kc2 < 2; ++kc2) {
                FragU vf;
                int vb = (nt * 16 + ln) * 72 + kc2 * 32 + half * 8;
                vf.q[0] = *(const uint4*)&Vt[vb];
                vf.q[1] = *(const uint4*)&Vt[vb + 16];
                o[nt] = __builtin_amdgcn_wmma_f32_16x16x32_bf16(
                    false, vf.v, false, pf[kc2].v, (short)0, o[nt], false, false);
            }
        }
    }

    // epilogue: per-lane scalar 1/l, 8 consecutive hs per lane -> vector stores
    {
        float inv = 1.0f / l;
        u16* yp = Y + ((long)(b * T + iq)) * CC + h * HS + 8 * half;
#pragma unroll
        for (int nt = 0; nt < 8; ++nt) {
            uint4 w;
            w.x = pack2bf(o[nt][0] * inv, o[nt][1] * inv);
            w.y = pack2bf(o[nt][2] * inv, o[nt][3] * inv);
            w.z = pack2bf(o[nt][4] * inv, o[nt][5] * inv);
            w.w = pack2bf(o[nt][6] * inv, o[nt][7] * inv);
            *(uint4*)(yp + nt * 16) = w;
        }
    }
}

// ---------------------------------------------------------------- launch
extern "C" void kernel_launch(void* const* d_in, const int* in_sizes, int n_in,
                              void* d_out, int out_size, void* d_ws, size_t ws_size,
                              hipStream_t stream) {
    constexpr int B = 2, T = 2048, C = 2048, KVC = 512;
    constexpr long BT = (long)B * T;

    const float* x  = (const float*)d_in[0];
    const float* Wq = (const float*)d_in[1];
    const float* Wk = (const float*)d_in[2];
    const float* Wv = (const float*)d_in[3];
    const float* Wo = (const float*)d_in[4];

    // Wq|Wk|Wv are laid out contiguously -> one packed (3072,2048) weight matrix.
    // Q|K|V outputs contiguous -> one packed (4096,3072) activation matrix.
    char* ws = (char*)d_ws;
    u16* xb   = (u16*)(ws);                        // 4096x2048
    u16* Wqkv = (u16*)(ws + 16777216);             // 3072x2048  (Wq | Wk | Wv)
    u16* Wkb  = (u16*)(ws + 25165824);             //   Wk part
    u16* Wvb  = (u16*)(ws + 27262976);             //   Wv part
    u16* Wob  = (u16*)(ws + 29360128);             // 2048x2048
    u16* QKVb = (u16*)(ws + 37748736);             // 4096x3072  (Q | K | V)
    u16* Yb   = (u16*)(ws + 62914560);             // 4096x2048

    cvt_f32_bf16<<<2048, 256, 0, stream>>>(x,  xb,   (int)(BT * C));
    cvt_f32_bf16<<<2048, 256, 0, stream>>>(Wq, Wqkv, C * C);
    cvt_f32_bf16<<<1024, 256, 0, stream>>>(Wk, Wkb,  KVC * C);
    cvt_f32_bf16<<<1024, 256, 0, stream>>>(Wv, Wvb,  KVC * C);
    cvt_f32_bf16<<<2048, 256, 0, stream>>>(Wo, Wob,  C * C);

    // fused QKV projection: x read once, 768 blocks
    gemm_xwt<0><<<dim3(3072 / 128, BT / 128), 256, 0, stream>>>(
        xb, Wqkv, nullptr, QKVb, (int)BT, 3072, C);

    attn_fwd<<<B * 16 * (T / 64), 128, 0, stream>>>(QKVb, Yb);

    gemm_xwt<1><<<dim3(C / 128, BT / 128), 256, 0, stream>>>(
        Yb, Wob, (float*)d_out, nullptr, (int)BT, C, C);
}
